// MultiLayerGNN_80487687127274
// MI455X (gfx1250) — matrix operand order repaired and checked by reading, not compiled, not run on hardware
//
#include <hip/hip_runtime.h>
#include <hip/hip_bf16.h>

typedef long long ll;
typedef unsigned short ushort_t;
typedef __attribute__((ext_vector_type(16))) __bf16 v16bf;
typedef __attribute__((ext_vector_type(8)))  float  v8f;

// ---------- helpers ----------
__device__ __forceinline__ unsigned short f2bf(float f) {
    unsigned u = __float_as_uint(f);
    unsigned r = u + 0x7fffu + ((u >> 16) & 1u);   // RNE
    return (unsigned short)(r >> 16);
}
__device__ __forceinline__ float bf2f(unsigned short b) {
    return __uint_as_float(((unsigned)b) << 16);
}
// order-preserving float->uint for atomicMax-based segment max
__device__ __forceinline__ unsigned encf(float f) {
    unsigned u = __float_as_uint(f);
    return (u & 0x80000000u) ? ~u : (u | 0x80000000u);
}
__device__ __forceinline__ float decf(unsigned u) {
    unsigned v = (u & 0x80000000u) ? (u & 0x7fffffffu) : ~u;
    return __uint_as_float(v);
}
#define NEG_INF_ENC 0x007FFFFFu   // encf(-inf)

// ---------- init kernels ----------
__global__ void zero_f(float* __restrict__ p, ll n) {
    ll t = (ll)blockIdx.x * blockDim.x + threadIdx.x;
    if (t < n) p[t] = 0.0f;
}
__global__ void init_u(unsigned* __restrict__ p, unsigned v, ll n) {
    ll t = (ll)blockIdx.x * blockDim.x + threadIdx.x;
    if (t < n) p[t] = v;
}

// ---------- W pre-swizzle: bf16 hi/lo in per-lane WMMA B-fragment order ----------
// Output layout: for chunk kc (K=kc*32..+31), tile nt, lane l:
//   halves [ ((kc*NT+nt)*32 + l)*16 + j ]  hold  W[kc*32 + kb/2 + j + (j>=8)*8][nt*16 + (l&15)]
//   with kb = (l>=16)*16  (ISA 7.12.2 B layout for 16x16x32 bf16)
__global__ void swizzle_W(const float* __restrict__ W,
                          ushort_t* __restrict__ Whi, ushort_t* __restrict__ Wlo,
                          int Ncols)
{
    int ntiles = Ncols >> 4;
    int total  = 4 * ntiles * 32;
    int t = blockIdx.x * blockDim.x + threadIdx.x;
    if (t >= total) return;
    int lane = t & 31;
    int grp  = t >> 5;              // kc*ntiles + nt
    int nt   = grp % ntiles;
    int kc   = grp / ntiles;
    int kb   = (lane >> 4) * 16;
    int n    = nt * 16 + (lane & 15);
    size_t base = (size_t)t * 16;
    for (int j = 0; j < 16; ++j) {
        int k = kc * 32 + (kb >> 1) + j + ((j >> 3) << 3);
        float v = W[(size_t)k * Ncols + n];
        unsigned short hi = f2bf(v);
        unsigned short lo = f2bf(v - bf2f(hi));
        Whi[base + j] = hi;
        Wlo[base + j] = lo;
    }
}

// ---------- GEMM: Y[M,NTILES*16] = X[M,128] @ W[128,NTILES*16], bf16x3 WMMA ----------
template <int NTILES>
__global__ __launch_bounds__(256) void gat_gemm(
    const float* __restrict__ X,
    const ushort_t* __restrict__ Whi, const ushort_t* __restrict__ Wlo,
    float* __restrict__ Y, int M)
{
    constexpr int NC = NTILES * 16;
    __shared__ __align__(16) ushort_t sAhi[128 * 32];
    __shared__ __align__(16) ushort_t sAlo[128 * 32];
    __shared__ __align__(16) ushort_t sWhi[NTILES * 512];
    __shared__ __align__(16) ushort_t sWlo[NTILES * 512];

    const int tid  = threadIdx.x;
    const int wave = tid >> 5;
    const int lane = tid & 31;
    const int rowBase = blockIdx.x * 128;

    v8f acc[NTILES];
    const v8f vzero = {0.f, 0.f, 0.f, 0.f, 0.f, 0.f, 0.f, 0.f};
    #pragma unroll
    for (int t = 0; t < NTILES; ++t) acc[t] = vzero;

    for (int kc = 0; kc < 4; ++kc) {
        // stage A chunk [128 rows][32 k] -> fragment-order bf16 hi/lo in LDS
        for (int i = tid; i < 128 * 32; i += 256) {
            int r = i >> 5, k = i & 31;
            int grow = rowBase + r;
            float v = 0.0f;
            if (grow < M) {
                v = X[(size_t)grow * 128 + (kc * 32 + k)];
                if (kc < 3)
                    __builtin_prefetch(&X[(size_t)grow * 128 + ((kc + 1) * 32 + k)], 0, 0);
            }
            unsigned short hi = f2bf(v);
            unsigned short lo = f2bf(v - bf2f(hi));
            // inverse of A fragment map (ISA 7.12.2, 16-bit A 16x32)
            int w  = r >> 4, mm = r & 15, g = k >> 3;
            int ln = mm + ((g & 1) << 4);
            int j  = k - (((g + 1) >> 1) << 3);
            int addr = ((w * 32 + ln) << 4) + j;
            sAhi[addr] = hi;
            sAlo[addr] = lo;
        }
        // stage W chunk: already fragment-ordered in global -> linear 16B copy
        {
            const uint4* srcH = (const uint4*)(Whi + (size_t)kc * NTILES * 512);
            const uint4* srcL = (const uint4*)(Wlo + (size_t)kc * NTILES * 512);
            uint4* dstH = (uint4*)sWhi;
            uint4* dstL = (uint4*)sWlo;
            for (int i = tid; i < NTILES * 64; i += 256) {   // NTILES*512 halves / 8 per uint4
                dstH[i] = srcH[i];
                dstL[i] = srcL[i];
            }
        }
        __syncthreads();

        // per-lane fragments are contiguous 32B -> single v16bf (2x ds_load_b128)
        v16bf ahi = *(const v16bf*)(sAhi + ((wave * 32 + lane) << 4));
        v16bf alo = *(const v16bf*)(sAlo + ((wave * 32 + lane) << 4));

        #pragma unroll
        for (int nt = 0; nt < NTILES; ++nt) {
            v16bf bhi = *(const v16bf*)(sWhi + ((nt * 32 + lane) << 4));
            v16bf blo = *(const v16bf*)(sWlo + ((nt * 32 + lane) << 4));
            // bf16x3: hi*hi + hi*lo + lo*hi, f32 accumulate
            acc[nt] = __builtin_amdgcn_wmma_f32_16x16x32_bf16(
                false, ahi, false, bhi, (short)0, acc[nt], false, false);
            acc[nt] = __builtin_amdgcn_wmma_f32_16x16x32_bf16(
                false, ahi, false, blo, (short)0, acc[nt], false, false);
            acc[nt] = __builtin_amdgcn_wmma_f32_16x16x32_bf16(
                false, alo, false, bhi, (short)0, acc[nt], false, false);
        }
        __syncthreads();
    }

    // C/D layout: VGPR r -> M = r + 8*(lane>=16), N = lane&15
    const int rowOff = rowBase + wave * 16 + ((lane >> 4) << 3);
    const int col    = lane & 15;
    #pragma unroll
    for (int nt = 0; nt < NTILES; ++nt) {
        #pragma unroll
        for (int r = 0; r < 8; ++r) {
            int grow = rowOff + r;
            if (grow < M)
                Y[(size_t)grow * NC + nt * 16 + col] = acc[nt][r];
        }
    }
}

// ---------- per-node attention dot products (H, C powers of two) ----------
__global__ void gat_dots(const float* __restrict__ h,
                         const float* __restrict__ as, const float* __restrict__ ad,
                         float* __restrict__ ss, float* __restrict__ dd,
                         int N, int logH, int logC)
{
    int H = 1 << logH, C = 1 << logC;
    int t = blockIdx.x * blockDim.x + threadIdx.x;
    if (t >= (N << logH)) return;
    int n  = t >> logH;
    int hd = t & (H - 1);
    int F  = H * C;
    const float* hp = h + (size_t)n * F + (hd << logC);
    const float* ap = as + (hd << logC);
    const float* bp = ad + (hd << logC);
    float s0 = 0.f, s1 = 0.f;
    for (int c = 0; c < C; ++c) { float v = hp[c]; s0 += v * ap[c]; s1 += v * bp[c]; }
    ss[t] = s0; dd[t] = s1;
}

// ---------- edge pass 1: segment max ----------
__global__ void gat_edge_max(const ll* __restrict__ ei, ll E, int N,
                             const float* __restrict__ ss, const float* __restrict__ dd,
                             unsigned* __restrict__ menc, int H)
{
    ll t = (ll)blockIdx.x * blockDim.x + threadIdx.x;
    ll ET = E + N;
    if (t >= ET) return;
    ll s, d;
    if (t < E) { s = ei[t]; d = ei[E + t]; } else { s = d = t - E; }
    for (int h = 0; h < H; ++h) {
        float e = ss[s * H + h] + dd[d * H + h];
        e = (e > 0.f) ? e : 0.2f * e;
        atomicMax(&menc[d * H + h], encf(e));
    }
}

// ---------- edge pass 2: segment sum of exp ----------
__global__ void gat_edge_sum(const ll* __restrict__ ei, ll E, int N,
                             const float* __restrict__ ss, const float* __restrict__ dd,
                             const unsigned* __restrict__ menc, float* __restrict__ sb, int H)
{
    ll t = (ll)blockIdx.x * blockDim.x + threadIdx.x;
    ll ET = E + N;
    if (t >= ET) return;
    ll s, d;
    if (t < E) { s = ei[t]; d = ei[E + t]; } else { s = d = t - E; }
    for (int h = 0; h < H; ++h) {
        float e = ss[s * H + h] + dd[d * H + h];
        e = (e > 0.f) ? e : 0.2f * e;
        float m = decf(menc[d * H + h]);
        atomicAdd(&sb[d * H + h], __expf(e - m));
    }
}

// ---------- edge pass 3: weighted aggregation (one wave per edge) ----------
__global__ __launch_bounds__(256) void gat_edge_agg(
    const ll* __restrict__ ei, ll E, int N,
    const float* __restrict__ h,
    const float* __restrict__ ss, const float* __restrict__ dd,
    const unsigned* __restrict__ menc, const float* __restrict__ sb,
    float* __restrict__ out, int H, int logC)
{
    int lane = threadIdx.x & 31;
    int wid  = threadIdx.x >> 5;
    ll edge = (ll)blockIdx.x * 8 + wid;
    ll ET = E + N;
    if (edge >= ET) return;
    ll s, d;
    if (edge < E) { s = ei[edge]; d = ei[E + edge]; } else { s = d = edge - E; }
    int F = H << logC;
    for (int f = lane; f < F; f += 32) {
        int hd = f >> logC;
        float e = ss[s * H + hd] + dd[d * H + hd];
        e = (e > 0.f) ? e : 0.2f * e;
        float m = decf(menc[d * H + hd]);
        float alpha = __expf(e - m) / (sb[d * H + hd] + 1e-16f);
        atomicAdd(&out[d * (ll)F + f], h[s * (ll)F + f] * alpha);
    }
}

// ---------- bias (+ optional ELU); F is a power of two ----------
__global__ void gat_finish(const float* __restrict__ in, const float* __restrict__ b,
                           float* __restrict__ out, ll total, int F, int do_elu)
{
    ll t = (ll)blockIdx.x * blockDim.x + threadIdx.x;
    if (t >= total) return;
    float v = in[t] + b[(int)(t & (F - 1))];
    if (do_elu) v = (v > 0.f) ? v : (__expf(v) - 1.0f);
    out[t] = v;
}

// ---------- host orchestration ----------
extern "C" void kernel_launch(void* const* d_in, const int* in_sizes, int n_in,
                              void* d_out, int out_size, void* d_ws, size_t ws_size,
                              hipStream_t stream)
{
    const float* x   = (const float*)d_in[0];
    const ll*    ei  = (const ll*)   d_in[1];
    const float* W1  = (const float*)d_in[2];
    const float* as1 = (const float*)d_in[3];
    const float* ad1 = (const float*)d_in[4];
    const float* b1  = (const float*)d_in[5];
    const float* W2  = (const float*)d_in[6];
    const float* as2 = (const float*)d_in[7];
    const float* ad2 = (const float*)d_in[8];
    const float* b2  = (const float*)d_in[9];
    const float* W3  = (const float*)d_in[10];
    const float* as3 = (const float*)d_in[11];
    const float* ad3 = (const float*)d_in[12];
    const float* b3  = (const float*)d_in[13];
    float* outp = (float*)d_out;

    const int N = in_sizes[0] / 128;
    const ll  E = (ll)in_sizes[1] / 2;
    const ll  ET = E + N;

    // workspace layout
    float* bufA = (float*)d_ws;                       // GEMM output h      [N*128]
    float* bufB = bufA + (size_t)N * 128;             // aggregation / act  [N*128]
    float* bufC = bufB + (size_t)N * 128;             // aggregation / act  [N*128]
    float* ss   = bufC + (size_t)N * 128;             // [N*4]
    float* dd   = ss   + (size_t)N * 4;               // [N*4]
    float* sb   = dd   + (size_t)N * 4;               // [N*4]
    unsigned*  menc = (unsigned*)(sb + (size_t)N * 4);   // [N*4]
    ushort_t*  whi  = (ushort_t*)(menc + (size_t)N * 4); // [16384]
    ushort_t*  wlo  = whi + 16384;                       // [16384]

    const int T = 256;
    const int mBlocks = (N + 127) / 128;
    const int edgeBlk = (int)((ET + T - 1) / T);
    const int aggBlk  = (int)((ET + 7) / 8);
    auto blocksFor = [](ll n) { return (int)((n + 255) / 256); };

    // ---------------- Layer 1: H=4, C=32, F=128, ELU ----------------
    swizzle_W<<<4, T, 0, stream>>>(W1, whi, wlo, 128);               // 1024 threads
    gat_gemm<8><<<mBlocks, T, 0, stream>>>(x, whi, wlo, bufA, N);
    gat_dots<<<blocksFor((ll)N * 4), T, 0, stream>>>(bufA, as1, ad1, ss, dd, N, 2, 5);
    init_u<<<blocksFor((ll)N * 4), T, 0, stream>>>(menc, NEG_INF_ENC, (ll)N * 4);
    zero_f<<<blocksFor((ll)N * 4), T, 0, stream>>>(sb, (ll)N * 4);
    zero_f<<<blocksFor((ll)N * 128), T, 0, stream>>>(bufB, (ll)N * 128);
    gat_edge_max<<<edgeBlk, T, 0, stream>>>(ei, E, N, ss, dd, menc, 4);
    gat_edge_sum<<<edgeBlk, T, 0, stream>>>(ei, E, N, ss, dd, menc, sb, 4);
    gat_edge_agg<<<aggBlk, T, 0, stream>>>(ei, E, N, bufA, ss, dd, menc, sb, bufB, 4, 5);
    gat_finish<<<blocksFor((ll)N * 128), T, 0, stream>>>(bufB, b1, bufB, (ll)N * 128, 128, 1);

    // ---------------- Layer 2: H=4, C=32, F=128, ELU ----------------
    swizzle_W<<<4, T, 0, stream>>>(W2, whi, wlo, 128);
    gat_gemm<8><<<mBlocks, T, 0, stream>>>(bufB, whi, wlo, bufA, N);
    gat_dots<<<blocksFor((ll)N * 4), T, 0, stream>>>(bufA, as2, ad2, ss, dd, N, 2, 5);
    init_u<<<blocksFor((ll)N * 4), T, 0, stream>>>(menc, NEG_INF_ENC, (ll)N * 4);
    zero_f<<<blocksFor((ll)N * 4), T, 0, stream>>>(sb, (ll)N * 4);
    zero_f<<<blocksFor((ll)N * 128), T, 0, stream>>>(bufC, (ll)N * 128);
    gat_edge_max<<<edgeBlk, T, 0, stream>>>(ei, E, N, ss, dd, menc, 4);
    gat_edge_sum<<<edgeBlk, T, 0, stream>>>(ei, E, N, ss, dd, menc, sb, 4);
    gat_edge_agg<<<aggBlk, T, 0, stream>>>(ei, E, N, bufA, ss, dd, menc, sb, bufC, 4, 5);
    gat_finish<<<blocksFor((ll)N * 128), T, 0, stream>>>(bufC, b2, bufC, (ll)N * 128, 128, 1);

    // ---------------- Layer 3: H=1, C=64, F=64, no ELU -> d_out ----------------
    swizzle_W<<<2, T, 0, stream>>>(W3, whi, wlo, 64);                // 512 threads
    gat_gemm<4><<<mBlocks, T, 0, stream>>>(bufC, whi, wlo, bufA, N);
    gat_dots<<<blocksFor((ll)N), T, 0, stream>>>(bufA, as3, ad3, ss, dd, N, 0, 6);
    init_u<<<blocksFor((ll)N), T, 0, stream>>>(menc, NEG_INF_ENC, (ll)N);
    zero_f<<<blocksFor((ll)N), T, 0, stream>>>(sb, (ll)N);
    zero_f<<<blocksFor((ll)N * 64), T, 0, stream>>>(bufB, (ll)N * 64);
    gat_edge_max<<<edgeBlk, T, 0, stream>>>(ei, E, N, ss, dd, menc, 1);
    gat_edge_sum<<<edgeBlk, T, 0, stream>>>(ei, E, N, ss, dd, menc, sb, 1);
    gat_edge_agg<<<aggBlk, T, 0, stream>>>(ei, E, N, bufA, ss, dd, menc, sb, bufB, 1, 6);
    gat_finish<<<blocksFor((ll)N * 64), T, 0, stream>>>(bufB, b3, outp, (ll)N * 64, 64, 0);
}
